// GlobalPointer_86681029968033
// MI455X (gfx1250) — matrix operand order repaired and checked by reading, not compiled, not run on hardware
//
#include <hip/hip_runtime.h>

// ---- types matching probe-confirmed WMMA/TDM builtin signatures ----
typedef __attribute__((ext_vector_type(16))) __bf16        v16bf;
typedef __attribute__((ext_vector_type(8)))  __bf16        v8bf;
typedef __attribute__((ext_vector_type(8)))  float         v8f;
typedef __attribute__((ext_vector_type(4)))  unsigned int  v4u;
typedef __attribute__((ext_vector_type(8)))  int           v8i_;
typedef __attribute__((ext_vector_type(4)))  int           v4i_;

#define BS   16
#define SEQ  512
#define HID  768
#define ENT  9
#define DIM  64
#define NOUT 1152                    // ENT * DIM * 2
#define TOTROWS (BS * ENT * SEQ)     // 73728 rows in the Qr/Kr tensors
#define BIGF 1000000000000.0f

// ---------------------------------------------------------------------------
// Kernel 0: W (768 x 1152, f32, row-major) -> Wt (1152 x 768, bf16, row-major)
// ---------------------------------------------------------------------------
__global__ void __launch_bounds__(256)
wt_transpose(const float* __restrict__ W, __bf16* __restrict__ Wt) {
    int w  = blockIdx.x * 256 + threadIdx.x;   // writes coalesced
    int n  = w / HID;
    int kk = w - n * HID;
    Wt[w] = (__bf16)W[(size_t)kk * NOUT + n];
}

// ---------------------------------------------------------------------------
// Kernel 1: Y = X @ W + b  (bf16 WMMA, f32 accum), fused RoPE epilogue.
// One wave computes a 16x64 output slab (A fragment reused across 4 N-tiles).
// ---------------------------------------------------------------------------
__global__ void __launch_bounds__(32)
proj_rope(const float* __restrict__ X, const __bf16* __restrict__ Wt,
          const float* __restrict__ bias,
          __bf16* __restrict__ Qr, __bf16* __restrict__ Kr) {
    const int lane = threadIdx.x;          // 0..31 (wave32)
    const int l16  = lane & 15;
    const int hi   = lane >> 4;
    const int m0   = blockIdx.x * 16;      // row tile over B*S = 8192
    const int n0   = blockIdx.y * 64;      // col slab over 1152

    v8f acc[4] = {};
    const float* xrow = X + (size_t)(m0 + l16) * HID + hi * 16;

    for (int k = 0; k < HID; k += 32) {
        // 16-bit A 16x32 layout: halfs 0..15 -> K=16*hi + 0..15 (contiguous, matches B)
        float4 a0 = *(const float4*)(xrow + k);
        float4 a1 = *(const float4*)(xrow + k + 4);
        float4 a2 = *(const float4*)(xrow + k + 8);
        float4 a3 = *(const float4*)(xrow + k + 12);
        v16bf a;
        a[0]  = (__bf16)a0.x; a[1]  = (__bf16)a0.y; a[2]  = (__bf16)a0.z; a[3]  = (__bf16)a0.w;
        a[4]  = (__bf16)a1.x; a[5]  = (__bf16)a1.y; a[6]  = (__bf16)a1.z; a[7]  = (__bf16)a1.w;
        a[8]  = (__bf16)a2.x; a[9]  = (__bf16)a2.y; a[10] = (__bf16)a2.z; a[11] = (__bf16)a2.w;
        a[12] = (__bf16)a3.x; a[13] = (__bf16)a3.y; a[14] = (__bf16)a3.z; a[15] = (__bf16)a3.w;

#pragma unroll
        for (int t = 0; t < 4; ++t) {
            const v16bf b = *(const v16bf*)(Wt + (size_t)(n0 + t * 16 + l16) * HID
                                               + k + hi * 16);
            acc[t] = __builtin_amdgcn_wmma_f32_16x16x32_bf16(
                         false, a, false, b, (short)0, acc[t], false, false);
        }
    }

    // Epilogue: bias + RoPE + bf16 store. C/D layout: VGPR r -> M = r + 8*hi, N = l16.
#pragma unroll
    for (int t = 0; t < 4; ++t) {
        const int   n    = n0 + t * 16 + l16;
        const float bn   = bias[n];
        const int   e    = n >> 7;
        const int   c128 = n & 127;
        const int   isK  = c128 >> 6;
        const int   d    = c128 & 63;
        const float freq = __powf(10000.0f, -(float)(d & ~1) * (1.0f / 64.0f));
        __bf16* dst = isK ? Kr : Qr;
#pragma unroll
        for (int r = 0; r < 8; ++r) {
            const int mg   = m0 + r + hi * 8;
            const int bidx = mg >> 9;
            const int s    = mg & 511;
            float val      = acc[t][r] + bn;
            float partner  = __shfl_xor(val, 1, 32);  // d XOR 1 lives in lane XOR 1
            const float ang = (float)s * freq;
            const float sn = __sinf(ang), cs = __cosf(ang);
            const float rot = (d & 1) ? (val * cs + partner * sn)
                                      : (val * cs - partner * sn);
            dst[(((size_t)bidx * ENT + e) * SEQ + s) * DIM + d] = (__bf16)rot;
        }
    }
}

// ---------------------------------------------------------------------------
// Tensor Data Mover: stage `rows` x 64 bf16 tile from global into LDS.
// Builds a D# per CDNA5 ISA Ch8: group0 {count=1, lds_addr, global_addr, type=2},
// group1 {data_size=2B, tensor_dim0=64, tensor_dim1=TOTROWS, tile=64 x rows,
// tensor_dim0_stride=64}; groups 2/3 zero (2-D tile). TDM ignores EXEC and is
// issued once per wave; completion via s_wait_tensorcnt.
// 6-arg builtin form (clang-23 / therock-10.0 toolchain, probe-confirmed):
//   (uint32x4, int32x8, int32x4, int32x4, int32x8, i32 cpol)
// ---------------------------------------------------------------------------
#if __has_builtin(__builtin_amdgcn_tensor_load_to_lds)
#define HAVE_TDM 1
__device__ __forceinline__ void tdm_load_rows(const __bf16* gsrc,
                                              unsigned lds_byte, int rows) {
    unsigned long long ga = (unsigned long long)(size_t)gsrc;
    v4u g0;
    g0[0] = 1u;                                            // count=1 (valid user D#)
    g0[1] = lds_byte;                                      // D#.lds_addr
    g0[2] = (unsigned)ga;                                  // global_addr[31:0]
    g0[3] = (unsigned)((ga >> 32) & 0x01FFFFFFu)           // global_addr[56:32]
          | (2u << 30);                                    // type = 2 ("image")
    v8i_ g1;
    g1[0] = (int)(1u << 16);                               // data_size = 2 bytes
    g1[1] = (int)((unsigned)DIM << 16);                    // tensor_dim0 = 64 (lo16)
    g1[2] = (int)(((unsigned)TOTROWS & 0xFFFFu) << 16);    // dim0 hi | tensor_dim1 lo16
    g1[3] = (int)(((unsigned)TOTROWS >> 16)                // tensor_dim1 hi16
          | ((unsigned)DIM << 16));                        // tile_dim0 = 64
    g1[4] = rows;                                          // tile_dim1 = rows, tile_dim2=0
    g1[5] = DIM;                                           // tensor_dim0_stride = 64
    g1[6] = 0;                                             // stride hi / dim1_stride lo
    g1[7] = 0;
    v4i_ gz4 = {0, 0, 0, 0};
    v8i_ gz8 = {0, 0, 0, 0, 0, 0, 0, 0};
    __builtin_amdgcn_tensor_load_to_lds(g0, g1, gz4, gz4, gz8, 0);
}
#else
#define HAVE_TDM 0
#endif

// ---------------------------------------------------------------------------
// Kernel 2: logits[b,e,m,n] = sum_d Qr[b,e,m,d] * Kr[b,e,n,d].
// 4-wave block computes a 16(m) x 64(n) slab. Q tile (16x64) + K slab (64x64)
// staged in LDS by the TDM (wave 0), then each wave runs two chained bf16
// WMMAs and the fused pad/tril/scale epilogue.
// ---------------------------------------------------------------------------
__global__ void __launch_bounds__(128)
qk_logits(const __bf16* __restrict__ Qr, const __bf16* __restrict__ Kr,
          const float* __restrict__ mask, float* __restrict__ out) {
    __shared__ __align__(16) __bf16 qtile[16 * DIM];   // 2 KB
    __shared__ __align__(16) __bf16 ktile[64 * DIM];   // 8 KB

    const int tid  = threadIdx.x;
    const int wave = tid >> 5;
    const int lane = tid & 31;
    const int l16  = lane & 15;
    const int hi   = lane >> 4;

    const int nb   = blockIdx.x;          // n slab (64 cols)
    const int mt   = blockIdx.y;          // m tile (16 rows)
    const int be   = blockIdx.z;          // b*ENT + e
    const int bidx = be / ENT;

    const __bf16* qsrc = Qr + ((size_t)be * SEQ + mt * 16) * DIM;
    const __bf16* ksrc = Kr + ((size_t)be * SEQ + nb * 64) * DIM;

#if HAVE_TDM
    if (wave == 0) {   // wave-uniform branch; TDM issued once per wave, EXEC ignored
        tdm_load_rows(qsrc, (unsigned)(size_t)&qtile[0], 16);
        tdm_load_rows(ksrc, (unsigned)(size_t)&ktile[0], 64);
        __builtin_amdgcn_s_wait_tensorcnt(0);
    }
#else
    for (int i = tid; i < (16 * DIM) / 8; i += 128)
        ((v8bf*)qtile)[i] = ((const v8bf*)qsrc)[i];
    for (int i = tid; i < (64 * DIM) / 8; i += 128)
        ((v8bf*)ktile)[i] = ((const v8bf*)ksrc)[i];
#endif
    __syncthreads();

    const __bf16* qbase = qtile + l16 * DIM;                 // A rows: M = l16
    const __bf16* kbase = ktile + (wave * 16 + l16) * DIM;   // B cols: N = l16 (per-wave subtile)

    v8f c = {};
#pragma unroll
    for (int kk = 0; kk < 2; ++kk) {
        // A fragment: K = 32*kk + 16*hi + {0..15} (two 16B ds loads)
        v8bf alo = *(const v8bf*)(qbase + kk * 32 + hi * 16);
        v8bf ahi = *(const v8bf*)(qbase + kk * 32 + hi * 16 + 8);
        v16bf a = __builtin_shufflevector(alo, ahi,
                      0, 1, 2, 3, 4, 5, 6, 7, 8, 9, 10, 11, 12, 13, 14, 15);
        // B fragment: contiguous 16 halfs, K = 32*kk + 16*hi + i (one 32B ds load)
        v16bf b = *(const v16bf*)(kbase + kk * 32 + hi * 16);
        c = __builtin_amdgcn_wmma_f32_16x16x32_bf16(
                false, a, false, b, (short)0, c, false, false);
    }

    const int   n   = nb * 64 + wave * 16 + l16;
    const float pad = mask[(size_t)bidx * SEQ + n];
    float* orow = out + ((size_t)be * SEQ + mt * 16) * SEQ + n;
#pragma unroll
    for (int r = 0; r < 8; ++r) {
        const int m = mt * 16 + r + hi * 8;
        float v = c[r] * pad - (1.0f - pad) * BIGF;     // padding mask
        if (m > n) v -= BIGF;                           // tril(-1) mask
        orow[(size_t)(r + hi * 8) * SEQ] = v * 0.125f;  // 1/sqrt(64)
    }
}

// ---------------------------------------------------------------------------
// Launch. ws layout: Wt bf16 @0 (1.7MB), Qr bf16 @2MB (9MB), Kr bf16 @11MB (9MB).
// ---------------------------------------------------------------------------
extern "C" void kernel_launch(void* const* d_in, const int* in_sizes, int n_in,
                              void* d_out, int out_size, void* d_ws, size_t ws_size,
                              hipStream_t stream) {
    const float* X    = (const float*)d_in[0];   // last_hidden_state (16,512,768)
    const float* mask = (const float*)d_in[1];   // attention_mask (16,512)
    const float* W    = (const float*)d_in[2];   // W_dense (768,1152)
    const float* bias = (const float*)d_in[3];   // b_dense (1152,)

    char*   ws  = (char*)d_ws;
    __bf16* Wt  = (__bf16*)(ws);
    __bf16* Qr  = (__bf16*)(ws + ((size_t)2  << 20));
    __bf16* Kr  = (__bf16*)(ws + ((size_t)11 << 20));
    float*  out = (float*)d_out;                 // (16,9,512,512) f32

    wt_transpose<<<dim3((NOUT * HID) / 256), 256, 0, stream>>>(W, Wt);
    proj_rope<<<dim3((BS * SEQ) / 16, NOUT / 64), 32, 0, stream>>>(X, Wt, bias, Qr, Kr);
    qk_logits<<<dim3(SEQ / 64, SEQ / 16, BS * ENT), 128, 0, stream>>>(Qr, Kr, mask, out);
}
